// NBODY_Transformer_50749333569846
// MI455X (gfx1250) — matrix operand order repaired and checked by reading, not compiled, not run on hardware
//
#include <hip/hip_runtime.h>
#include <hip/hip_bf16.h>
#include <math.h>

typedef __attribute__((ext_vector_type(16))) _Float16 v16h;
typedef __attribute__((ext_vector_type(8)))  float    v8f;

#define N_NODES 5
#define N_EDGES 20
#define SEQ     25
#define CIN     3
#define DM      7
#define DMODEL  56
#define DFF     224
#define NHEADS  8
#define HDIM    7
#define NLAYERS 4

// ---------------- LDS arena (float units). All matrix buffers padded so every
// fragment load is unconditional and in-bounds; K-padding zeros always live on
// the A side (or in the masked K^T fragment), so B-side garbage is annihilated.
//   XB : residual x  f32 [32][64]                  2048 floats
//   HB : LN out      f16 [32][64] (cols56-63 = 0)  1024 floats
//   QB/KB/VB/OB : Q,K,V,attn-out f16 [32][64]      1024 floats each
//   AB : attn probs  f16 [8][32][32] (cols25-31=0) 4096 floats (heads 4-7 overlap SC)
//   SC : scores f32 [8][32][32]  (aliases FFN mid f16 [32][224] + GA staging)
#define XB_OFF 0
#define HB_OFF 2048
#define QB_OFF 3072
#define KB_OFF 4096
#define VB_OFF 5120
#define OB_OFF 6144
#define AB_OFF 7168
#define SC_OFF 9216
#define ARENA_F (SC_OFF + 8192)   // 17408 floats = 69632 B

// ---------------- WMMA fragment helpers (CDNA5 wave32 layouts) --------------

// A fragment 16x32 f16 from padded f16 LDS. Unconditional.
// VGPR r holds K pair kb,kb+1, kb = 2r + 16*(lane>=16).
__device__ __forceinline__ v16h load_a16(const _Float16* S, int ld, int m0,
                                         int koff, int lane) {
  const _Float16* row = S + (m0 + (lane & 15)) * ld + koff;
  int hi = lane >> 4;
  v16h a;
#pragma unroll
  for (int r = 0; r < 8; ++r) {
    int kb = 2 * r + 16 * hi;
    a[2 * r]     = row[kb];
    a[2 * r + 1] = row[kb + 1];
  }
  return a;
}

// B fragment 32x16 f16 from f16 LDS, B[k][n] = S[(koff+k)*ld + n0+n]. Unconditional.
__device__ __forceinline__ v16h load_b16(const _Float16* S, int ld, int koff,
                                         int n0, int lane) {
  int n  = n0 + (lane & 15);
  int kb = koff + (lane >> 4) * 16;
  v16h b;
#pragma unroll
  for (int j = 0; j < 16; ++j) b[j] = S[(kb + j) * ld + n];
  return b;
}

// B fragment for K^T in the score matmul: B[k][n] = S[(n0+n)*ld + koff+k],
// valid only for k < HDIM; unconditional clamped load + value select (cndmask).
__device__ __forceinline__ v16h load_bT16_h(const _Float16* S, int ld, int koff,
                                            int n0, int lane) {
  const _Float16* row = S + (n0 + (lane & 15)) * ld + koff;
  int kb = (lane >> 4) * 16;
  v16h b;
#pragma unroll
  for (int j = 0; j < 16; ++j) {
    int k = kb + j;
    _Float16 x = row[k < HDIM ? k : 0];
    b[j] = (k < HDIM) ? x : (_Float16)0.f;
  }
  return b;
}

// B fragment from global f32 weights: clamped addresses, NO value masking
// (A-side K padding is zero). Converts to f16.
__device__ __forceinline__ v16h load_bg(const float* W, int ld, int koff, int n0,
                                        int kmax, int nmax, int lane) {
  int n = n0 + (lane & 15);
  n = (n < nmax) ? n : (nmax - 1);
  int kb = koff + (lane >> 4) * 16;
  v16h b;
#pragma unroll
  for (int j = 0; j < 16; ++j) {
    int k = kb + j;
    k = (k < kmax) ? k : (kmax - 1);
    b[j] = (_Float16)W[(size_t)k * ld + n];
  }
  return b;
}

// C fragment store, f16, unconditional (padded destination).
__device__ __forceinline__ void store_c16(_Float16* D, int ld, int m0, int n0,
                                          v8f c, int lane) {
  int n  = n0 + (lane & 15);
  int hi = lane >> 4;
#pragma unroll
  for (int r = 0; r < 8; ++r) D[(m0 + r + 8 * hi) * ld + n] = (_Float16)c[r];
}

#define WMMA(a, b, c) \
  __builtin_amdgcn_wmma_f32_16x16x32_f16(false, (a), false, (b), (short)0, (c), false, false)

// --------------------------------- kernel -----------------------------------

__global__ __launch_bounds__(256)
void nbody_tf_fused(const float* __restrict__ nodes, const float* __restrict__ edges,
                    const float* __restrict__ mask,
                    const float* __restrict__ W_in, const float* __restrict__ b_in,
                    const float* __restrict__ W_gp, const float* __restrict__ b_gp,
                    const float* __restrict__ Wq, const float* __restrict__ Wk,
                    const float* __restrict__ Wv, const float* __restrict__ Wo,
                    const float* __restrict__ ln1g, const float* __restrict__ ln1b,
                    const float* __restrict__ ln2g, const float* __restrict__ ln2b,
                    const float* __restrict__ W1, const float* __restrict__ b1,
                    const float* __restrict__ W2, const float* __restrict__ b2,
                    float* __restrict__ out) {
  __shared__ float arena[ARENA_F];

  float*     xb = arena + XB_OFF;                    // f32 [32][64]
  _Float16*  hb = (_Float16*)(arena + HB_OFF);       // f16 [32][64]
  _Float16*  qb = (_Float16*)(arena + QB_OFF);       // f16 [32][64]
  _Float16*  kbuf = (_Float16*)(arena + KB_OFF);     // f16 [32][64]
  _Float16*  vb = (_Float16*)(arena + VB_OFF);       // f16 [32][64]
  _Float16*  obuf = (_Float16*)(arena + OB_OFF);     // f16 [32][64]
  _Float16*  ab = (_Float16*)(arena + AB_OFF);       // f16 [8][32][32]
  float*     sc = arena + SC_OFF;                    // f32 [8][32][32]
  _Float16*  fb = (_Float16*)(arena + SC_OFF);       // f16 [32][224] (aliases sc)
  float*     ga = arena + SC_OFF;                    // f32 [25][14][8] (aliases sc)

  const int tid  = threadIdx.x;
  const int lane = tid & 31;
  const int wave = tid >> 5;
  const int bidx = blockIdx.x;

  const int GRADE[8] = {0, 1, 1, 1, 2, 2, 2, 3};
  const int BL[8]    = {0, 1, 2, 4, 3, 5, 6, 7};  // blade order == idx map

  // One-time: zero obuf column padding (cols 56..63, rows 0..24). It is the A
  // matrix of the Wo matmul, and per-head stores never touch those columns.
  if (tid < SEQ * 8) obuf[(tid >> 3) * 64 + 56 + (tid & 7)] = (_Float16)0.f;

  // ---------------- Phase A1: mv_linear(W_in) + geometric product -----------
  for (int item = tid; item < SEQ * DM; item += blockDim.x) {
    int t = item / DM, n = item % DM;
    const float* src = (t < N_NODES)
        ? nodes + ((size_t)bidx * N_NODES + t) * CIN * 8
        : edges + ((size_t)bidx * N_EDGES + (t - N_NODES)) * CIN * 8;
    float mv[8];
#pragma unroll
    for (int i = 0; i < 8; ++i) {
      float acc = 0.f;
#pragma unroll
      for (int m = 0; m < CIN; ++m)
        acc += src[m * 8 + i] * W_in[(n * CIN + m) * 4 + GRADE[i]];
      mv[i] = acc;
    }
    mv[0] += b_in[n];

    float gp[8] = {0.f, 0.f, 0.f, 0.f, 0.f, 0.f, 0.f, 0.f};
#pragma unroll
    for (int ai = 0; ai < 8; ++ai) {
#pragma unroll
      for (int bi = 0; bi < 8; ++bi) {
        int ba = BL[ai], bbv = BL[bi];
        int s = 0, tt = ba >> 1;
        while (tt) { s += __popc(tt & bbv); tt >>= 1; }
        int j = BL[ba ^ bbv];
        float sg = (s & 1) ? -1.f : 1.f;
        gp[j] += sg * mv[ai] * mv[bi];
      }
    }
    float* d0 = ga + (t * 14 + n) * 8;
    float* d1 = ga + (t * 14 + DM + n) * 8;
#pragma unroll
    for (int i = 0; i < 8; ++i) { d0[i] = mv[i]; d1[i] = gp[i]; }
  }
  __syncthreads();

  // ---------------- Phase A2: mv_linear(W_gp) -> x --------------------------
  for (int item = tid; item < SEQ * DM; item += blockDim.x) {
    int t = item / DM, n = item % DM;
    const float* mg = ga + t * 14 * 8;
#pragma unroll
    for (int i = 0; i < 8; ++i) {
      float acc = (i == 0) ? b_gp[n] : 0.f;
      for (int m = 0; m < 14; ++m)
        acc += mg[m * 8 + i] * W_gp[(n * 14 + m) * 4 + GRADE[i]];
      xb[t * 64 + n * 8 + i] = acc;
    }
  }
  __syncthreads();

  // ---------------- Phase B: transformer layers -----------------------------
  for (int l = 0; l < NLAYERS; ++l) {
    const float* wq  = Wq + (size_t)l * DMODEL * DMODEL;
    const float* wk  = Wk + (size_t)l * DMODEL * DMODEL;
    const float* wv  = Wv + (size_t)l * DMODEL * DMODEL;
    const float* wo  = Wo + (size_t)l * DMODEL * DMODEL;
    const float* w1  = W1 + (size_t)l * DMODEL * DFF;
    const float* bb1 = b1 + (size_t)l * DFF;
    const float* w2  = W2 + (size_t)l * DFF * DMODEL;
    const float* bb2 = b2 + (size_t)l * DMODEL;

    // B1: LayerNorm-1 -> hb (writes zero padding in cols 56..63)
    if (tid < SEQ) {
      const float* xr = xb + tid * 64;
      float m = 0.f;
      for (int d = 0; d < DMODEL; ++d) m += xr[d];
      m *= (1.f / DMODEL);
      float v = 0.f;
      for (int d = 0; d < DMODEL; ++d) { float z = xr[d] - m; v += z * z; }
      v *= (1.f / DMODEL);
      float rs = rsqrtf(v + 1e-5f);
      const float* g = ln1g + (size_t)l * DMODEL;
      const float* b = ln1b + (size_t)l * DMODEL;
      _Float16* hr = hb + tid * 64;
      for (int d = 0; d < DMODEL; ++d) hr[d] = (_Float16)((xr[d] - m) * rs * g[d] + b[d]);
      for (int d = DMODEL; d < 64; ++d) hr[d] = (_Float16)0.f;
    }
    __syncthreads();

    // B2: Q,K,V = h @ W  (one 16x16 tile per wave, unconditional stores)
    {
      int mi = wave & 1, ni = wave >> 1;
      const float* Ws[3] = {wq, wk, wv};
      const int    Oo[3] = {QB_OFF, KB_OFF, VB_OFF};
      for (int mm = 0; mm < 3; ++mm) {
        v8f c = {0.f, 0.f, 0.f, 0.f, 0.f, 0.f, 0.f, 0.f};
#pragma unroll
        for (int k0 = 0; k0 < DMODEL; k0 += 32) {
          v16h a = load_a16(hb, 64, mi * 16, k0, lane);
          v16h b = load_bg(Ws[mm], DMODEL, k0, ni * 16, DMODEL, DMODEL, lane);
          c = WMMA(a, b, c);
        }
        store_c16((_Float16*)(arena + Oo[mm]), 64, mi * 16, ni * 16, c, lane);
      }
    }
    __syncthreads();

    // B3: scores = Q K^T / sqrt(hd) + mask  (one head per wave)
    {
      int h = wave;
      const float rscale = 0.3779644730092272f;  // 1/sqrt(7)
      const float* mk = mask + (size_t)bidx * SEQ * SEQ;
#pragma unroll
      for (int mi = 0; mi < 2; ++mi)
#pragma unroll
        for (int ni = 0; ni < 2; ++ni) {
          v8f c = {0.f, 0.f, 0.f, 0.f, 0.f, 0.f, 0.f, 0.f};
          v16h a = load_a16(qb, 64, mi * 16, h * HDIM, lane);
          v16h b = load_bT16_h(kbuf, 64, h * HDIM, ni * 16, lane);
          c = WMMA(a, b, c);
          int n   = ni * 16 + (lane & 15);
          int hi2 = lane >> 4;
          int nc  = (n < SEQ) ? n : (SEQ - 1);
#pragma unroll
          for (int r = 0; r < 8; ++r) {
            int m  = mi * 16 + r + 8 * hi2;
            int mc = (m < SEQ) ? m : (SEQ - 1);
            sc[(h * 32 + m) * 32 + n] = c[r] * rscale + mk[mc * SEQ + nc];
          }
        }
    }
    __syncthreads();

    // B4: softmax over keys; write attn probs as f16 with zero K-padding
    {
      int h = tid >> 5, q = tid & 31;
      if (q < SEQ) {
        const float* row = sc + (h * 32 + q) * 32;
        _Float16* arow = ab + (h * 32 + q) * 32;
        float mx = -1e30f;
        for (int k = 0; k < SEQ; ++k) mx = fmaxf(mx, row[k]);
        float s = 0.f;
        float e[SEQ];
        for (int k = 0; k < SEQ; ++k) { e[k] = __expf(row[k] - mx); s += e[k]; }
        float inv = 1.f / s;
        for (int k = 0; k < SEQ; ++k) arow[k] = (_Float16)(e[k] * inv);
        for (int k = SEQ; k < 32; ++k) arow[k] = (_Float16)0.f;
      }
    }
    __syncthreads();

    // B5: O = attn @ V  (per head; store only the 7 head columns)
    {
      int h = wave;
#pragma unroll
      for (int mi = 0; mi < 2; ++mi) {
        v8f c = {0.f, 0.f, 0.f, 0.f, 0.f, 0.f, 0.f, 0.f};
        v16h a = load_a16(ab + h * 32 * 32, 32, mi * 16, 0, lane);
        v16h b = load_b16(vb, 64, 0, h * HDIM, lane);
        c = WMMA(a, b, c);
        int nl = lane & 15;
        int hi2 = lane >> 4;
        if (nl < HDIM) {
          int n = h * HDIM + nl;
#pragma unroll
          for (int r = 0; r < 8; ++r)
            obuf[(mi * 16 + r + 8 * hi2) * 64 + n] = (_Float16)c[r];
        }
      }
    }
    __syncthreads();

    // B6: x += O @ Wo  (unconditional += into padded x)
    {
      int mi = wave & 1, ni = wave >> 1;
      v8f c = {0.f, 0.f, 0.f, 0.f, 0.f, 0.f, 0.f, 0.f};
#pragma unroll
      for (int k0 = 0; k0 < DMODEL; k0 += 32) {
        v16h a = load_a16(obuf, 64, mi * 16, k0, lane);
        v16h b = load_bg(wo, DMODEL, k0, ni * 16, DMODEL, DMODEL, lane);
        c = WMMA(a, b, c);
      }
      int n = ni * 16 + (lane & 15);
      int hi2 = lane >> 4;
#pragma unroll
      for (int r = 0; r < 8; ++r)
        xb[(mi * 16 + r + 8 * hi2) * 64 + n] += c[r];
    }
    __syncthreads();

    // B7: LayerNorm-2 -> hb
    if (tid < SEQ) {
      const float* xr = xb + tid * 64;
      float m = 0.f;
      for (int d = 0; d < DMODEL; ++d) m += xr[d];
      m *= (1.f / DMODEL);
      float v = 0.f;
      for (int d = 0; d < DMODEL; ++d) { float z = xr[d] - m; v += z * z; }
      v *= (1.f / DMODEL);
      float rs = rsqrtf(v + 1e-5f);
      const float* g = ln2g + (size_t)l * DMODEL;
      const float* b = ln2b + (size_t)l * DMODEL;
      _Float16* hr = hb + tid * 64;
      for (int d = 0; d < DMODEL; ++d) hr[d] = (_Float16)((xr[d] - m) * rs * g[d] + b[d]);
      for (int d = DMODEL; d < 64; ++d) hr[d] = (_Float16)0.f;
    }
    __syncthreads();

    // B8: ff = gelu(h @ W1 + b1) -> fb (f16 [32][224], unconditional stores)
    {
      for (int t = wave; t < 28; t += 8) {
        int mi = t & 1, ni = t >> 1;
        v8f c = {0.f, 0.f, 0.f, 0.f, 0.f, 0.f, 0.f, 0.f};
#pragma unroll
        for (int k0 = 0; k0 < DMODEL; k0 += 32) {
          v16h a = load_a16(hb, 64, mi * 16, k0, lane);
          v16h b = load_bg(w1, DFF, k0, ni * 16, DMODEL, DFF, lane);
          c = WMMA(a, b, c);
        }
        int n = ni * 16 + (lane & 15);
        int hi2 = lane >> 4;
        float bv = bb1[n];
#pragma unroll
        for (int r = 0; r < 8; ++r) {
          float z = c[r] + bv;
          float u = 0.7978845608028654f * (z + 0.044715f * z * z * z);
          fb[(mi * 16 + r + 8 * hi2) * DFF + n] = (_Float16)(0.5f * z * (1.f + tanhf(u)));
        }
      }
    }
    __syncthreads();

    // B9: x += ff @ W2 + b2  (K = 224 exactly, no padding anywhere)
    {
      int mi = wave & 1, ni = wave >> 1;
      v8f c = {0.f, 0.f, 0.f, 0.f, 0.f, 0.f, 0.f, 0.f};
      for (int k0 = 0; k0 < DFF; k0 += 32) {
        v16h a = load_a16(fb, DFF, mi * 16, k0, lane);
        v16h b = load_bg(w2, DMODEL, k0, ni * 16, DFF, DMODEL, lane);
        c = WMMA(a, b, c);
      }
      int n = ni * 16 + (lane & 15);
      int hi2 = lane >> 4;
      float bv = bb2[(n < DMODEL) ? n : (DMODEL - 1)];
#pragma unroll
      for (int r = 0; r < 8; ++r)
        xb[(mi * 16 + r + 8 * hi2) * 64 + n] += c[r] + bv;
    }
    __syncthreads();
  }

  // ---------------- Phase C: select out[:, :5, channel 1, :] ----------------
  if (tid < N_NODES * 8) {
    int t = tid >> 3, i = tid & 7;
    out[((size_t)bidx * N_NODES + t) * 8 + i] = xb[t * 64 + 8 + i];
  }
}

// ------------------------------- launcher -----------------------------------

extern "C" void kernel_launch(void* const* d_in, const int* in_sizes, int n_in,
                              void* d_out, int out_size, void* d_ws, size_t ws_size,
                              hipStream_t stream) {
  (void)n_in; (void)out_size; (void)d_ws; (void)ws_size;
  const float* nodes = (const float*)d_in[0];
  const float* edges = (const float*)d_in[1];
  const float* mask  = (const float*)d_in[2];
  /* d_in[3] = batch_size (device scalar, unused; B derived from sizes) */
  const float* W_in = (const float*)d_in[4];
  const float* b_in = (const float*)d_in[5];
  const float* W_gp = (const float*)d_in[6];
  const float* b_gp = (const float*)d_in[7];
  const float* Wq   = (const float*)d_in[8];
  const float* Wk   = (const float*)d_in[9];
  const float* Wv   = (const float*)d_in[10];
  const float* Wo   = (const float*)d_in[11];
  const float* ln1g = (const float*)d_in[12];
  const float* ln1b = (const float*)d_in[13];
  const float* ln2g = (const float*)d_in[14];
  const float* ln2b = (const float*)d_in[15];
  const float* W1   = (const float*)d_in[16];
  const float* b1   = (const float*)d_in[17];
  const float* W2   = (const float*)d_in[18];
  const float* b2   = (const float*)d_in[19];

  int B = in_sizes[0] / (N_NODES * CIN * 8);
  if (B <= 0) return;

  hipLaunchKernelGGL(nbody_tf_fused, dim3(B), dim3(256), 0, stream,
                     nodes, edges, mask, W_in, b_in, W_gp, b_gp,
                     Wq, Wk, Wv, Wo, ln1g, ln1b, ln2g, ln2b,
                     W1, b1, W2, b2, (float*)d_out);
}
